// GraphConvolution_13692355740361
// MI455X (gfx1250) — matrix operand order repaired and checked by reading, not compiled, not run on hardware
//
#include <hip/hip_runtime.h>
#include <hip/hip_bf16.h>

#define NN 50000
#define NE 800000
#define DD 256

typedef __attribute__((ext_vector_type(16))) __bf16 v16bf;
typedef __attribute__((ext_vector_type(8)))  __bf16 v8bf;
typedef __attribute__((ext_vector_type(8)))  float  v8f;

__device__ __forceinline__ unsigned short f2bf(float f) {
    union { float f; unsigned u; } a; a.f = f;
    unsigned u = a.u;
    unsigned r = u + 0x7FFFu + ((u >> 16) & 1u);   // round-to-nearest-even
    return (unsigned short)(r >> 16);
}

// ---------------------------------------------------------------------------
// Kernel 0: fp32 -> bf16 bulk convert (one-time; amortizes across all tiles).
// ---------------------------------------------------------------------------
__global__ __launch_bounds__(256) void gcn_cvt(const float* __restrict__ in,
                                               unsigned short* __restrict__ outb,
                                               int n) {
    int base = (blockIdx.x * 256 + threadIdx.x) * 8;
    if (base >= n) return;
    float4 a = *(const float4*)(in + base);
    float4 b = *(const float4*)(in + base + 4);
    union { unsigned short u[8]; uint4 q; } o;
    o.u[0] = f2bf(a.x); o.u[1] = f2bf(a.y); o.u[2] = f2bf(a.z); o.u[3] = f2bf(a.w);
    o.u[4] = f2bf(b.x); o.u[5] = f2bf(b.y); o.u[6] = f2bf(b.z); o.u[7] = f2bf(b.w);
    *(uint4*)(outb + base) = o.q;
}

// ---------------------------------------------------------------------------
// Kernel 1: out = x + bias   (epilogue pre-fold; atomics accumulate on top)
// ---------------------------------------------------------------------------
__global__ __launch_bounds__(256) void gcn_init(const float* __restrict__ X,
                                                const float* __restrict__ bias,
                                                float* __restrict__ out) {
    size_t base = ((size_t)blockIdx.x * blockDim.x + threadIdx.x) * 4;
    if (base >= (size_t)NN * DD) return;
    float4 x4 = *(const float4*)(X + base);
    float4 b4 = *(const float4*)(bias + (base & (DD - 1)));
    float4 o;
    o.x = x4.x + b4.x; o.y = x4.y + b4.y;
    o.z = x4.z + b4.z; o.w = x4.w + b4.w;
    *(float4*)(out + base) = o;
}

// ---------------------------------------------------------------------------
// Kernel 2: support = X @ W, bf16 operands, f32 accumulate.
// One wave computes a 32x64 block: 2 m-tiles x 4 n-tiles. Per k-step:
// 4 b128 A-loads + 64 u16 B-loads (all issued before the WMMA group -> partial
// s_wait_loadcnt) feed 8 v_wmma_f32_16x16x32_bf16. 0.5 b128 + 8 u16 loads per WMMA.
//
// Fragment layouts (ISA 7.12.2, wave32):
//   A 16x32 bf16 : lane l%16 = row M; kb = 8*(lane>=16); K={kb..kb+7}U{16+kb..}
//   B 32x16 bf16 : lane l%16 = col N; same K set as A (strided gathers).
//   C/D 16x16 f32: VGPR r -> M = r + 8*(lane>=16), N = lane%16.
// ---------------------------------------------------------------------------
__global__ __launch_bounds__(256) void gcn_gemm(const __bf16* __restrict__ Xb,
                                                const __bf16* __restrict__ Wb,
                                                float* __restrict__ support) {
    const int MG   = (NN + 31) / 32;                 // 1563 m-groups of 32 rows
    const int lane = threadIdx.x & 31;
    const int wave = threadIdx.x >> 5;
    const int task = blockIdx.x * 8 + wave;          // [0, MG*4)
    if (task >= MG * 4) return;
    const int mg   = task >> 2;                      // 32-row group  [0,1563)
    const int nb   = (task & 3) * 64;                // col base {0,64,128,192}
    const int half = lane >> 4;
    const int l    = lane & 15;

    const int  m0     = mg * 32 + l;                 // tile0 row (always valid)
    int        m1     = m0 + 16;                     // tile1 row (may be OOB)
    const bool tile1  = (mg * 32 + 16) < NN;         // whole tile1 valid?
    if (m1 >= NN) m1 = NN - 1;                       // clamp loads in-bounds

    const __bf16* arow0 = Xb + (size_t)m0 * DD;
    const __bf16* arow1 = Xb + (size_t)m1 * DD;

    v8f acc[2][4] = {{{}, {}, {}, {}}, {{}, {}, {}, {}}};

    #pragma unroll
    for (int k0 = 0; k0 < DD; k0 += 32) {
        union { v16bf v; v8bf h[2]; } A0, A1, B[4];
        const int ka = k0 + half * 8;
        A0.h[0] = *(const v8bf*)(arow0 + ka);
        A0.h[1] = *(const v8bf*)(arow0 + ka + 16);
        A1.h[0] = *(const v8bf*)(arow1 + ka);
        A1.h[1] = *(const v8bf*)(arow1 + ka + 16);

        const __bf16* brow = Wb + (size_t)(k0 + half * 8) * DD + nb + l;
        #pragma unroll
        for (int nt = 0; nt < 4; ++nt) {
            #pragma unroll
            for (int s = 0; s < 8; ++s) {
                B[nt].h[0][s] = brow[(size_t)s * DD + nt * 16];
                B[nt].h[1][s] = brow[(size_t)(s + 16) * DD + nt * 16];
            }
        }

        #pragma unroll
        for (int nt = 0; nt < 4; ++nt) {
            acc[0][nt] = __builtin_amdgcn_wmma_f32_16x16x32_bf16(
                false, A0.v, false, B[nt].v, (short)0, acc[0][nt], false, false);
            acc[1][nt] = __builtin_amdgcn_wmma_f32_16x16x32_bf16(
                false, A1.v, false, B[nt].v, (short)0, acc[1][nt], false, false);
        }
    }

    // tile0 stores (always valid)
    #pragma unroll
    for (int nt = 0; nt < 4; ++nt) {
        float* orow = support + (size_t)(mg * 32 + half * 8) * DD + nb + nt * 16 + l;
        #pragma unroll
        for (int r = 0; r < 8; ++r) orow[(size_t)r * DD] = acc[0][nt][r];
    }
    // tile1 stores (skipped only for the final partial 32-row group)
    if (tile1) {
        #pragma unroll
        for (int nt = 0; nt < 4; ++nt) {
            float* orow = support + (size_t)(mg * 32 + 16 + half * 8) * DD + nb + nt * 16 + l;
            #pragma unroll
            for (int r = 0; r < 8; ++r) orow[(size_t)r * DD] = acc[1][nt][r];
        }
    }
}

// ---------------------------------------------------------------------------
// Kernel 3: edge-parallel scatter: out[row] += val * support[col]
// 64 lanes per edge, float4 per lane; L2-resident gathers + global_atomic_add_f32.
// ---------------------------------------------------------------------------
__global__ __launch_bounds__(256) void gcn_agg(const float* __restrict__ support,
                                               const int*   __restrict__ erow,
                                               const int*   __restrict__ ecol,
                                               const float* __restrict__ eval,
                                               float* __restrict__ out) {
    const int g = threadIdx.x >> 6;
    const int l = threadIdx.x & 63;
    const int e = blockIdx.x * 4 + g;
    if (e >= NE) return;
    const int   r = erow[e];
    const int   c = ecol[e];
    const float v = eval[e];
    const float4 m = ((const float4*)(support + (size_t)c * DD))[l];
    float* dst = out + (size_t)r * DD + l * 4;
    unsafeAtomicAdd(dst + 0, m.x * v);
    unsafeAtomicAdd(dst + 1, m.y * v);
    unsafeAtomicAdd(dst + 2, m.z * v);
    unsafeAtomicAdd(dst + 3, m.w * v);
}

// ---------------------------------------------------------------------------
// Kernel 4: out = relu(out)
// ---------------------------------------------------------------------------
__global__ __launch_bounds__(256) void gcn_relu(float* __restrict__ out) {
    size_t base = ((size_t)blockIdx.x * blockDim.x + threadIdx.x) * 4;
    if (base >= (size_t)NN * DD) return;
    float4 o = *(float4*)(out + base);
    o.x = fmaxf(o.x, 0.f); o.y = fmaxf(o.y, 0.f);
    o.z = fmaxf(o.z, 0.f); o.w = fmaxf(o.w, 0.f);
    *(float4*)(out + base) = o;
}

extern "C" void kernel_launch(void* const* d_in, const int* in_sizes, int n_in,
                              void* d_out, int out_size, void* d_ws, size_t ws_size,
                              hipStream_t stream) {
    const float* x    = (const float*)d_in[0];
    const int*   erow = (const int*)  d_in[1];
    const int*   ecol = (const int*)  d_in[2];
    const float* ev   = (const float*)d_in[3];
    const float* w    = (const float*)d_in[4];
    const float* bias = (const float*)d_in[5];
    float* out = (float*)d_out;

    // Workspace layout: [support f32 51.2MB][Xb bf16 25.6MB][Wb bf16 128KB]
    char* ws = (char*)d_ws;
    float*          support = (float*)ws;
    unsigned short* Xb      = (unsigned short*)(ws + (size_t)NN * DD * 4);
    unsigned short* Wb      = (unsigned short*)(ws + (size_t)NN * DD * 4 + (size_t)NN * DD * 2);

    const int nx = NN * DD;          // 12.8M
    const int nw = DD * DD;          // 64K
    gcn_cvt <<<(nx / 8 + 255) / 256, 256, 0, stream>>>(x, Xb, nx);
    gcn_cvt <<<(nw / 8 + 255) / 256, 256, 0, stream>>>(w, Wb, nw);
    gcn_init<<<(nx / 4 + 255) / 256, 256, 0, stream>>>(x, bias, out);

    const int tasks = ((NN + 31) / 32) * 4;          // 6252 wave tasks
    gcn_gemm<<<(tasks + 7) / 8, 256, 0, stream>>>((const __bf16*)Xb, (const __bf16*)Wb, support);

    gcn_agg <<<NE / 4, 256, 0, stream>>>(support, erow, ecol, ev, out);
    gcn_relu<<<(nx / 4 + 255) / 256, 256, 0, stream>>>(out);
}
